// ActivationDiffusionBlock_78632261255882
// MI455X (gfx1250) — matrix operand order repaired and checked by reading, not compiled, not run on hardware
//
#include <hip/hip_runtime.h>
#include <hip/hip_bf16.h>
#include <math.h>

// ---------------------------------------------------------------------------
// Problem constants
// ---------------------------------------------------------------------------
constexpr int PH    = 14;
constexpr int PW    = 14;
constexpr int NPTS  = PH * PW;      // 196 patches
constexpr int NP    = 224;          // padded: 14*16, and 7*32 (f16 WMMA K-steps)
constexpr int NTILE = NP / 16;      // 14
constexpr int KSTEP = NP / 32;      // 7
constexpr int CCH   = 768;          // channels
constexpr int BATCH = 256;
constexpr int NSLOT = NPTS * 5;     // self + 4 neighbors per patch = 980
constexpr int BLK   = 512;          // 16 waves of 32

constexpr int CHUNK  = 16;                  // channels staged per step
constexpr int NCHUNK = CCH / CHUNK;         // 48
constexpr int STGF   = NPTS * CHUNK;        // 3136 floats per stage buffer
constexpr int AISSUE = 7;                   // ceil(STGF / BLK): async copies/thread

// LDS layout (bytes)
constexpr int XA_OFF = 0;
constexpr int XB_OFF = NP * NP * 2;               // 100352
constexpr int YT_OFF = 2 * NP * NP * 2;           // 200704
constexpr int CF_OFF = 3 * NP * NP * 2;           // 301056 (dots, then coefs)
constexpr int A5_OFF = CF_OFF + NSLOT * 4;        // +3920
constexpr int RN_OFF = A5_OFF + NP * 4;
constexpr int SMEM_BYTES = RN_OFF + NP * 4;       // ~306.8 KB < 320 KB/WGP

typedef _Float16 v16h __attribute__((ext_vector_type(16)));
typedef _Float16 v8h  __attribute__((ext_vector_type(8)));
typedef float    v8f  __attribute__((ext_vector_type(8)));

// ---------------------------------------------------------------------------
// Async global->LDS (CDNA5: GLOBAL_LOAD_ASYNC_TO_LDS_B32, ASYNCcnt tracked),
// guarded so the kernel still compiles on toolchains without the builtins.
// Builtin signature uses int-typed pointers: (int AS1*, int AS3*, Iimm, Iimm).
// ---------------------------------------------------------------------------
#if __has_builtin(__builtin_amdgcn_global_load_async_to_lds_b32) && \
    __has_builtin(__builtin_amdgcn_s_wait_asynccnt)
#define USE_ASYNC_LDS 1
typedef __attribute__((address_space(1))) int* gint_p;
typedef __attribute__((address_space(3))) int* lint_p;
#else
#define USE_ASYNC_LDS 0
#endif

__device__ __forceinline__ void async_copy_f32(const float* g, float* l) {
#if USE_ASYNC_LDS
    __builtin_amdgcn_global_load_async_to_lds_b32((gint_p)g, (lint_p)l, 0, 0);
#else
    *l = *g;
#endif
}
__device__ __forceinline__ void wait_async_prev() {   // prev chunk done, next in flight
#if USE_ASYNC_LDS
    __builtin_amdgcn_s_wait_asynccnt(AISSUE);
#endif
}
__device__ __forceinline__ void wait_async_all() {
#if USE_ASYNC_LDS
    __builtin_amdgcn_s_wait_asynccnt(0);
#endif
}

__device__ __forceinline__ int nbr_of(int n, int j) {
    int r = n / PW, c = n % PW;
    switch (j) {
        case 0: return n;
        case 1: return (r > 0)      ? n - PW : -1;
        case 2: return (r < PH - 1) ? n + PW : -1;
        case 3: return (c > 0)      ? n - 1  : -1;
        default:return (c < PW - 1) ? n + 1  : -1;
    }
}
__device__ __forceinline__ float deg_of(int n) {
    int r = n / PW, c = n % PW;
    return (float)((r > 0) + (r < PH - 1) + (c > 0) + (c < PW - 1));
}

// ---------------------------------------------------------------------------
// Kernel 1: one workgroup per batch. Everything except the final S-scaling
// lives in LDS: stencil Gram -> L coefs -> Newton-Schulz (WMMA f16) -> F_map.
// ---------------------------------------------------------------------------
__global__ __launch_bounds__(BLK)
void graph_diffuse_kernel(const float* __restrict__ S,
                          const float* __restrict__ attn,
                          const float* __restrict__ lmbd,
                          const float* __restrict__ beta,
                          float* __restrict__ Fout) {
    extern __shared__ char smem[];
    _Float16* Xbuf0 = (_Float16*)(smem + XA_OFF);
    _Float16* Xbuf1 = (_Float16*)(smem + XB_OFF);
    _Float16* Yt    = (_Float16*)(smem + YT_OFF);
    float*    coef  = (float*)(smem + CF_OFF);     // dots, then L coefficients
    float*    a5    = (float*)(smem + A5_OFF);
    float*    rn    = (float*)(smem + RN_OFF);
    // double-buffered transposed stage [n][16 channels], inside Xbuf1 region
    float*    stg0  = (float*)(smem + XB_OFF);     // 3136 floats
    float*    stg1  = stg0 + STGF;                 // 3136 floats (= 25088 B total)

    const int tid = threadIdx.x;
    const int b   = blockIdx.x;
    const float lm = lmbd[0];
    const float bt = beta[0];
    const float* Sb = S + (size_t)b * CCH * NPTS;

    // attention vector, zero-padded to NP
    for (int n = tid; n < NP; n += BLK)
        a5[n] = (n < NPTS) ? attn[b * NPTS + n] * 5.0f : 0.0f;

    // ---- Phase 1: stencil dot products (only E entries L ever touches) ----
    // Stage chunk cc of S transposed into stg[n*16 + c]; async double-buffer.
    const int s0 = tid, s1 = tid + BLK;
    const int n0 = s0 / 5, m0 = nbr_of(n0, s0 % 5);
    const int n1 = (s1 < NSLOT) ? s1 / 5 : 0;
    const int m1 = (s1 < NSLOT) ? nbr_of(n1, s1 % 5) : -1;
    float acc0 = 0.0f, acc1 = 0.0f;

    // issue exactly AISSUE copies per thread (wrap duplicates are idempotent)
    auto issue_chunk = [&](int cc, float* dst) {
        const float* src = Sb + (size_t)cc * CHUNK * NPTS;
        #pragma unroll
        for (int k = 0; k < AISSUE; ++k) {
            int idx = tid + k * BLK;
            if (idx >= STGF) idx -= STGF;
            int c = idx / NPTS, n = idx - c * NPTS;
            async_copy_f32(src + (size_t)c * NPTS + n, dst + n * CHUNK + c);
        }
    };

    issue_chunk(0, stg0);
    for (int cc = 0; cc < NCHUNK; ++cc) {
        const float* stg = (cc & 1) ? stg1 : stg0;
        if (cc + 1 < NCHUNK) {
            issue_chunk(cc + 1, (cc & 1) ? stg0 : stg1);
            wait_async_prev();          // in-order: chunk cc now complete
        } else {
            wait_async_all();
        }
        __syncthreads();
        {   // float4 dot accumulation over this chunk's 16 channels
            const float4* pa = (const float4*)(stg + n0 * CHUNK);
            if (m0 >= 0) {
                const float4* pb = (const float4*)(stg + m0 * CHUNK);
                #pragma unroll
                for (int q = 0; q < 4; ++q) {
                    float4 xa = pa[q], xb = pb[q];
                    acc0 += xa.x * xb.x + xa.y * xb.y + xa.z * xb.z + xa.w * xb.w;
                }
            }
            if (m1 >= 0) {
                const float4* pc = (const float4*)(stg + n1 * CHUNK);
                const float4* pd = (const float4*)(stg + m1 * CHUNK);
                #pragma unroll
                for (int q = 0; q < 4; ++q) {
                    float4 xa = pc[q], xb = pd[q];
                    acc1 += xa.x * xb.x + xa.y * xb.y + xa.z * xb.z + xa.w * xb.w;
                }
            }
        }
        __syncthreads();                // readers done before next overwrite
    }
    coef[s0] = acc0;                    // raw dots
    if (s1 < NSLOT) coef[s1] = acc1;
    __syncthreads();

    // ---- Phase 2: inverse norms, then L stencil coefficients (in place) ---
    for (int n = tid; n < NP; n += BLK)
        rn[n] = (n < NPTS) ? 1.0f / fmaxf(sqrtf(coef[n * 5]), 1e-12f) : 0.0f;
    __syncthreads();
    for (int s = tid; s < NSLOT; s += BLK) {
        int n = s / 5, j = s - n * 5;
        int m = nbr_of(n, j);
        float cf = 0.0f;
        if (m >= 0) {
            float Ev  = (coef[s] * rn[n] * rn[m] + 1.0f) * 0.5f;
            float dma = (j == 0) ? deg_of(n) : -1.0f;
            cf = dma * (lm * Ev - 1.0f) + ((j == 0) ? 1e-6f : 0.0f);
        }
        coef[s] = cf;                   // each thread rewrites its own slot
    }
    __syncthreads();

    // ---- Phase 3: X0 = 0.01 * L^T (sparse scatter into zeroed f16 X) ------
    for (int i = tid; i < NP * NP; i += BLK) Xbuf0[i] = (_Float16)0.0f;
    __syncthreads();
    for (int s = tid; s < NSLOT; s += BLK) {
        int n = s / 5, j = s - n * 5;
        int m = nbr_of(n, j);
        if (m >= 0) Xbuf0[m * NP + n] = (_Float16)(0.01f * coef[s]);
    }
    __syncthreads();

    // ---- Phase 4: Newton-Schulz, X <- X (2I - L X), L applied as stencil --
    const int  wave = tid >> 5;
    const int  lane = tid & 31;
    const int  lrow = lane & 15;
    const bool hi   = lane >= 16;

    _Float16* Xc = Xbuf0;
    _Float16* Xn = Xbuf1;
    for (int it = 0; it < 4; ++it) {
        // Yt[m][n] = (2I - L*Xc)[n][m], vectorized over 8 consecutive m
        for (int i = tid; i < NP * NP / 8; i += BLK) {
            int n  = i / (NP / 8);
            int mq = (i - n * (NP / 8)) * 8;
            float y[8];
            #pragma unroll
            for (int q = 0; q < 8; ++q) y[q] = (mq + q == n) ? 2.0f : 0.0f;
            if (n < NPTS) {
                #pragma unroll
                for (int j = 0; j < 5; ++j) {
                    int k = nbr_of(n, j);
                    if (k >= 0) {
                        float cf = coef[n * 5 + j];
                        v8h xv = *(const v8h*)(Xc + k * NP + mq);
                        #pragma unroll
                        for (int q = 0; q < 8; ++q) y[q] -= cf * (float)xv[q];
                    }
                }
            }
            #pragma unroll
            for (int q = 0; q < 8; ++q)
                Yt[(mq + q) * NP + n] = (_Float16)y[q];
        }
        __syncthreads();

        // Xn = Xc @ Y via v_wmma_f32_16x16x32_f16 (software-pipelined K loop)
        auto loadA = [&](int mt, int ks) -> v16h {
            const _Float16* ap = Xc + (mt * 16 + lrow) * NP + ks * 32 + (hi ? 8 : 0);
            v8h alo = *(const v8h*)ap;
            v8h ahi = *(const v8h*)(ap + 16);
            return __builtin_shufflevector(alo, ahi,
                0, 1, 2, 3, 4, 5, 6, 7, 8, 9, 10, 11, 12, 13, 14, 15);
        };
        auto loadB = [&](int nt, int ks) -> v16h {
            const _Float16* bp = Yt + (nt * 16 + lrow) * NP + ks * 32 + (hi ? 8 : 0);
            v8h blo = *(const v8h*)bp;
            v8h bhi = *(const v8h*)(bp + 16);
            return __builtin_shufflevector(blo, bhi,
                0, 1, 2, 3, 4, 5, 6, 7, 8, 9, 10, 11, 12, 13, 14, 15);
        };
        for (int t = wave; t < NTILE * NTILE; t += BLK / 32) {
            int mt = t / NTILE, nt = t - mt * NTILE;
            v8f acc = {};
            v16h af = loadA(mt, 0);
            v16h bf = loadB(nt, 0);
            #pragma unroll
            for (int ks = 0; ks < KSTEP; ++ks) {
                v16h afn = af, bfn = bf;
                if (ks + 1 < KSTEP) {            // prefetch next fragments
                    afn = loadA(mt, ks + 1);
                    bfn = loadB(nt, ks + 1);
                }
                acc = __builtin_amdgcn_wmma_f32_16x16x32_f16(
                    false, af, false, bf, (short)0, acc, false, false);
                af = afn; bf = bfn;
            }
            #pragma unroll
            for (int v = 0; v < 8; ++v) {
                int M  = mt * 16 + (hi ? 8 : 0) + v;
                int Nc = nt * 16 + lrow;
                Xn[M * NP + Nc] = (_Float16)acc[v];
            }
        }
        __syncthreads();
        _Float16* tp = Xc; Xc = Xn; Xn = tp;   // 4 iters -> result back in Xbuf0
    }

    // ---- Phase 5: Fd = X @ a5 (v8h rows), tanh-shrinkage, emit F_map ------
    for (int n = tid; n < NPTS; n += BLK) {
        const v8h* row = (const v8h*)(Xc + n * NP);
        float fd = 0.0f;
        for (int mb = 0; mb < NP / 8; ++mb) {
            v8h xv = row[mb];
            const float* av = a5 + mb * 8;
            #pragma unroll
            for (int q = 0; q < 8; ++q) fd += (float)xv[q] * av[q];
        }
        float fdiff = bt * (fd - tanhf(fd / (bt + 1e-8f)));
        Fout[b * NPTS + n] = fdiff;
    }
}

// ---------------------------------------------------------------------------
// Kernel 2: pure-bandwidth pass  S_new = S * (1 + F_map)   (float4 vectors;
// NPTS % 4 == 0 so a float4 never straddles a patch row)
// ---------------------------------------------------------------------------
__global__ __launch_bounds__(256)
void scale_kernel(const float* __restrict__ S,
                  const float* __restrict__ Fmap,
                  float* __restrict__ out) {
    size_t i4   = (size_t)blockIdx.x * 256 + threadIdx.x;
    size_t base = i4 * 4;
    constexpr size_t TOT = (size_t)BATCH * CCH * NPTS;
    if (base >= TOT) return;
    int    n0 = (int)(base % NPTS);
    size_t bc = base / NPTS;            // b*C + c
    size_t bb = bc / CCH;
    const float4 s = *(const float4*)(S + base);
    const float4 f = *(const float4*)(Fmap + bb * NPTS + n0);
    float4 r;
    r.x = s.x * (1.0f + f.x);
    r.y = s.y * (1.0f + f.y);
    r.z = s.z * (1.0f + f.z);
    r.w = s.w * (1.0f + f.w);
    *(float4*)(out + base) = r;
}

// ---------------------------------------------------------------------------
extern "C" void kernel_launch(void* const* d_in, const int* in_sizes, int n_in,
                              void* d_out, int out_size, void* d_ws, size_t ws_size,
                              hipStream_t stream) {
    const float* S    = (const float*)d_in[0];
    const float* attn = (const float*)d_in[1];
    const float* lmbd = (const float*)d_in[2];
    const float* beta = (const float*)d_in[3];

    float* out_S = (float*)d_out;                                   // [B,C,H,W]
    float* out_F = out_S + (size_t)BATCH * CCH * NPTS;              // [B,1,H,W]

    graph_diffuse_kernel<<<BATCH, BLK, SMEM_BYTES, stream>>>(
        S, attn, lmbd, beta, out_F);

    constexpr size_t TOT   = (size_t)BATCH * CCH * NPTS;
    constexpr int    NBLK2 = (int)((TOT / 4 + 255) / 256);          // 37632
    scale_kernel<<<NBLK2, 256, 0, stream>>>(S, out_F, out_S);
}
